// RNTN_12824772346472
// MI455X (gfx1250) — matrix operand order repaired and checked by reading, not compiled, not run on hardware
//
#include <hip/hip_runtime.h>

// ---------------------------------------------------------------------------
// RNTN forward, MI455X (gfx1250, wave32, bf16 WMMA, async global->LDS).
//
// tp[b,n] = sum_{i<=j} p_i p_j V'[i,j,n],  V' = V + V^T(ij) folded (upper tri)
// + linear term W_w folded in as 8 extra K-chunks.  K-chunks with jc < i/32
// are all-zero and skipped -> 1152 of 2048 chunks remain (44% saved).
// ---------------------------------------------------------------------------

typedef __attribute__((ext_vector_type(16))) __bf16 v16bf;
typedef __attribute__((ext_vector_type(8)))  __bf16 v8bf;
typedef __attribute__((ext_vector_type(8)))  float  v8f;
typedef int v4i __attribute__((vector_size(16)));   // matches builtin proto

#define DDIM      128
#define D2        256
#define NCHUNK_V  2048                  // 65536/32
#define NCHUNK    (NCHUNK_V + 8)        // + W_w chunks
#define NFRAG     (NCHUNK * 8)
#define BLOCK     128                   // 4 waves: 2 row-groups x 2 col-groups
#define WG_ROWS   64
#define STAGE_OFF   32768               // phrase/h slab size (64*256*2 B)
#define STAGE_BYTES 65536               // one i-step of B fragments
#define LDS_BYTES   (STAGE_OFF + 2 * STAGE_BYTES)   // 160 KB (dynamic)

#if defined(__AMDGCN__) && __has_builtin(__builtin_amdgcn_global_load_async_to_lds_b128)
#define USE_ASYNC 1
#else
#define USE_ASYNC 0
#endif

__device__ __forceinline__ void cp16(char* l, const char* g) {
#if USE_ASYNC
  __builtin_amdgcn_global_load_async_to_lds_b128(
      (__attribute__((address_space(1))) v4i*)g,
      (__attribute__((address_space(3))) v4i*)l, 0, 0);
#else
  *(int4*)l = *(const int4*)g;          // synchronous fallback
#endif
}

__device__ __forceinline__ void wait_async0() {
#if USE_ASYNC
#if __has_builtin(__builtin_amdgcn_s_wait_asynccnt)
  __builtin_amdgcn_s_wait_asynccnt(0);
#else
  asm volatile("s_wait_asynccnt 0x0" ::: "memory");
#endif
#endif
}

// Fragment layout: per frag (1024 B): lane's elems 0..7 at lane*16, elems
// 8..15 at 512+lane*16  ->  ds_load_b128 hits every LDS bank exactly twice.
__device__ __forceinline__ v16bf ld_frag(const __bf16* fb, int lane) {
  v8bf lo = *(const v8bf*)(fb + lane * 8);
  v8bf hi = *(const v8bf*)(fb + 256 + lane * 8);
  return __builtin_shufflevector(lo, hi,
           0,1,2,3,4,5,6,7,8,9,10,11,12,13,14,15);
}

// ---------------------------------------------------------------------------
// Pack V' (symmetry-folded, bf16) and W_w^T into WMMA-B fragment layout.
// frag (c,t): K = c*32..+31, N = t*16..+15; lane l<16 -> K 0..15, col t*16+l;
// lane l>=16 -> K 16..31.  Split-half storage as described above.
// ---------------------------------------------------------------------------
__global__ __launch_bounds__(256) void rntn_pack_vb(
    const float* __restrict__ V, const float* __restrict__ Ww,
    __bf16* __restrict__ vb)
{
  int g    = blockIdx.x * 256 + threadIdx.x;   // [0, NFRAG*32)
  int lane = g & 31;
  int frag = g >> 5;
  int c    = frag >> 3;
  int t    = frag & 7;
  int n    = t * 16 + (lane & 15);
  int kb   = (lane >> 4) << 4;                 // 0 or 16
  v8bf lo, hi;
  if (c < NCHUNK_V) {
    int i  = c >> 3;
    int jc = c & 7;
#pragma unroll
    for (int e = 0; e < 16; ++e) {
      int j = jc * 32 + kb + e;
      float v;
      if (j > i)       v = V[(size_t)(i * D2 + j) * DDIM + n] +
                           V[(size_t)(j * D2 + i) * DDIM + n];
      else if (j == i) v = V[(size_t)(i * D2 + i) * DDIM + n];
      else             v = 0.f;
      if (e < 8) lo[e] = (__bf16)v; else hi[e - 8] = (__bf16)v;
    }
  } else {
    int j0 = (c - NCHUNK_V) * 32 + kb;
#pragma unroll
    for (int e = 0; e < 16; ++e) {
      float v = Ww[(size_t)n * D2 + j0 + e];
      if (e < 8) lo[e] = (__bf16)v; else hi[e - 8] = (__bf16)v;
    }
  }
  __bf16* dst = vb + (size_t)frag * 512 + lane * 8;
  *(v8bf*)dst         = lo;
  *(v8bf*)(dst + 256) = hi;
}

// ---------------------------------------------------------------------------
// Main kernel: 128 threads = 4 waves in a 2x2 grid.
// Wave (wr,wc): rows wr*32..+31 (2 row tiles, pj in regs), cols wc*64..+63.
// B fragments staged in double-buffered LDS by async copies, shared by all.
// ---------------------------------------------------------------------------
__global__ __launch_bounds__(BLOCK) void rntn_main(
    const float* __restrict__ left,  const float* __restrict__ right,
    const __bf16* __restrict__ vb,
    const float* __restrict__ Wb,    const float* __restrict__ Wsw,
    const float* __restrict__ Wsb,   float* __restrict__ out)
{
  extern __shared__ char lds[];
  const int tid  = threadIdx.x;
  const int wave = tid >> 5;
  const int lane = tid & 31;
  const int r    = lane & 15;
  const int hi   = lane >> 4;
  const int wr   = wave >> 1;          // row group 0..1
  const int wc   = wave & 1;           // col group 0..1

  __bf16* ph   = (__bf16*)lds;         // [64][256] bf16 phrase
  char* stage  = lds + STAGE_OFF;      // 2 x 64KB B staging
  const int rowWG = blockIdx.x * WG_ROWS;

  // Prologue: start async copy of i=0 B slab into buffer 0.
  for (int off = tid * 16; off < STAGE_BYTES; off += BLOCK * 16)
    cp16(stage + off, (const char*)vb + off);

  // Cooperative phrase fill (concat(left,right) -> bf16 LDS).
  for (int idx = tid; idx < WG_ROWS * D2; idx += BLOCK) {
    int row = rowWG + (idx >> 8);
    int col = idx & 255;
    float v = (col < DDIM) ? left[(size_t)row * DDIM + col]
                           : right[(size_t)row * DDIM + col - DDIM];
    ph[idx] = (__bf16)v;
  }
  __syncthreads();

  // pj fragments (phrase rows in 16x32 A layout) for this wave's 2 row tiles.
  const int lr0 = wr * 32 + r;
  const int lr1 = lr0 + 16;
  v16bf pj[2][8];
#pragma unroll
  for (int jc = 0; jc < 8; ++jc) {
    int j0 = jc * 32 + (hi << 4);
    v8bf a0 = *(const v8bf*)(ph + lr0 * D2 + j0);
    v8bf a1 = *(const v8bf*)(ph + lr0 * D2 + j0 + 8);
    pj[0][jc] = __builtin_shufflevector(a0, a1, 0,1,2,3,4,5,6,7,8,9,10,11,12,13,14,15);
    v8bf b0 = *(const v8bf*)(ph + lr1 * D2 + j0);
    v8bf b1 = *(const v8bf*)(ph + lr1 * D2 + j0 + 8);
    pj[1][jc] = __builtin_shufflevector(b0, b1, 0,1,2,3,4,5,6,7,8,9,10,11,12,13,14,15);
  }

  v8f acc[2][4] = {};

  for (int i = 0; i < D2; ++i) {
    wait_async0();                      // own share of buf[i&1] copies done
    __syncthreads();                    // all shares done; prev buf reads done
    if (i + 1 < D2) {                   // issue copies for next i-step
      int jn = (i + 1) >> 5;            // skip all-zero chunks (symmetry fold)
      const char* g = (const char*)vb + (size_t)(i + 1) * STAGE_BYTES;
      char* dst = stage + ((i + 1) & 1) * STAGE_BYTES;
      for (int off = jn * 8192 + tid * 16; off < STAGE_BYTES; off += BLOCK * 16)
        cp16(dst + off, g + off);
    }
    const char* buf = stage + (i & 1) * STAGE_BYTES;
    const int jc0 = i >> 5;             // uniform: lower-triangle chunks skip
    __bf16 s0 = ph[lr0 * D2 + i];
    __bf16 s1 = ph[lr1 * D2 + i];
#pragma unroll
    for (int jc = 0; jc < 8; ++jc) {
      if (jc >= jc0) {
        const __bf16* cb = (const __bf16*)(buf + jc * 8192) + (wc * 4) * 512;
        // Preload all 4 B fragments first: one clause of 8 ds_load_b128,
        // single s_wait_dscnt, then 8 back-to-back WMMAs.
        v16bf b[4];
#pragma unroll
        for (int u = 0; u < 4; ++u) b[u] = ld_frag(cb + u * 512, lane);
        v16bf a0 = pj[0][jc] * s0;      // v_pk_mul_bf16 x8 each (covers loads)
        v16bf a1 = pj[1][jc] * s1;
#pragma unroll
        for (int u = 0; u < 4; ++u) {
          acc[0][u] = __builtin_amdgcn_wmma_f32_16x16x32_bf16(
              false, a0, false, b[u], (short)0, acc[0][u], false, false);
          acc[1][u] = __builtin_amdgcn_wmma_f32_16x16x32_bf16(
              false, a1, false, b[u], (short)0, acc[1][u], false, false);
        }
      }
    }
  }

  // Folded linear term: A = phrase, B = packed W_w^T (direct from global).
#pragma unroll
  for (int jc = 0; jc < 8; ++jc) {
    v16bf b[4];
#pragma unroll
    for (int u = 0; u < 4; ++u) {
      const __bf16* fb = vb + (size_t)((NCHUNK_V + jc) * 8 + wc * 4 + u) * 512;
      b[u] = ld_frag(fb, lane);
    }
#pragma unroll
    for (int u = 0; u < 4; ++u) {
      acc[0][u] = __builtin_amdgcn_wmma_f32_16x16x32_bf16(
          false, pj[0][jc], false, b[u], (short)0, acc[0][u], false, false);
      acc[1][u] = __builtin_amdgcn_wmma_f32_16x16x32_bf16(
          false, pj[1][jc], false, b[u], (short)0, acc[1][u], false, false);
    }
  }

  __syncthreads();                      // phrase reads done; retype slab
  float* hS = (float*)lds;              // [64][128] f32 h

  // h = tanh(tp + lin + W_b); C layout: elem v -> row v+hi*8, col base+r.
#pragma unroll
  for (int q = 0; q < 2; ++q)
#pragma unroll
    for (int u = 0; u < 4; ++u) {
      int n = (wc * 4 + u) * 16 + r;
      float bias = Wb[n];
#pragma unroll
      for (int v = 0; v < 8; ++v)
        hS[(wr * 32 + q * 16 + v + hi * 8) * DDIM + n] = tanhf(acc[q][u][v] + bias);
    }
  __syncthreads();                      // rows gather cols from both wc waves

  // Classifier + log_softmax: one row per thread (threads 0..63).
  if (tid < WG_ROWS) {
    float lg[5];
#pragma unroll
    for (int cc = 0; cc < 5; ++cc) lg[cc] = Wsb[cc];
    for (int j = 0; j < DDIM; ++j) {
      int jj = (j + (tid & 31) * 4) & (DDIM - 1);   // stagger LDS banks
      float hv = hS[tid * DDIM + jj];
#pragma unroll
      for (int cc = 0; cc < 5; ++cc)
        lg[cc] = fmaf(hv, Wsw[cc * DDIM + jj], lg[cc]);
    }
    float m = lg[0];
#pragma unroll
    for (int cc = 1; cc < 5; ++cc) m = fmaxf(m, lg[cc]);
    float ssum = 0.f;
#pragma unroll
    for (int cc = 0; cc < 5; ++cc) ssum += expf(lg[cc] - m);
    float lse = m + logf(ssum);
    int row = rowWG + tid;
#pragma unroll
    for (int cc = 0; cc < 5; ++cc)
      out[(size_t)row * 5 + cc] = lg[cc] - lse;
  }
}

// ---------------------------------------------------------------------------
extern "C" void kernel_launch(void* const* d_in, const int* in_sizes, int n_in,
                              void* d_out, int out_size, void* d_ws, size_t ws_size,
                              hipStream_t stream) {
  const float* left = (const float*)d_in[0];   // [8192,128]
  const float* right = (const float*)d_in[1];  // [8192,128]
  const float* V    = (const float*)d_in[2];   // [256,256,128]
  const float* Ww   = (const float*)d_in[3];   // [128,256]
  const float* Wb   = (const float*)d_in[4];   // [128]
  const float* Wsw  = (const float*)d_in[5];   // [5,128]
  const float* Wsb  = (const float*)d_in[6];   // [5]
  float* out        = (float*)d_out;           // [8192,5]

  __bf16* vb = (__bf16*)d_ws;                  // NFRAG*1024 B ~= 16.1 MiB

  rntn_pack_vb<<<NFRAG * 32 / 256, 256, 0, stream>>>(V, Ww, vb);
  rntn_main<<<8192 / WG_ROWS, BLOCK, LDS_BYTES, stream>>>(
      left, right, vb, Wb, Wsw, Wsb, out);
}
